// MultiHeadAttention_14242111554176
// MI455X (gfx1250) — compile-verified, matrix operand order checked by reading, not run on hardware
//
#include <hip/hip_runtime.h>

// ---------------------------------------------------------------------------
// MHA forward for MI455X (gfx1250, wave32, WMMA bf16 16x16x32 f32-accum,
// async global->LDS double buffering).
// B=2, S=2048, D=1024, H=16, Da=64
// ---------------------------------------------------------------------------

#define B_   2
#define S_   2048
#define D_   1024
#define H_   16
#define DA_  64
#define BH_  (B_ * H_)
#define M_   (B_ * S_)          // 4096 rows in the projection GEMMs

typedef __attribute__((ext_vector_type(16))) __bf16 v16bf;
typedef __attribute__((ext_vector_type(8)))  __bf16 v8bf;
typedef __attribute__((ext_vector_type(4)))  __bf16 v4bf;
typedef __attribute__((ext_vector_type(8)))  float  v8f;
typedef __attribute__((ext_vector_type(4)))  float  v4f;
typedef __attribute__((ext_vector_type(4)))  int    v4i;

#if defined(__gfx1250__) && \
    __has_builtin(__builtin_amdgcn_global_load_async_to_lds_b128) && \
    __has_builtin(__builtin_amdgcn_s_wait_asynccnt)
#define USE_ASYNC 1
#else
#define USE_ASYNC 0
#endif

__device__ inline void wait_async() {
#if USE_ASYNC
  __builtin_amdgcn_s_wait_asynccnt(0);
#endif
}

// one 16-byte chunk: global -> LDS (async when available)
__device__ inline void copy16(const __bf16* g, __bf16* l) {
#if USE_ASYNC
  __builtin_amdgcn_global_load_async_to_lds_b128((v4i*)g, (v4i*)l, 0, 0);
#else
  *(v8bf*)l = *(const v8bf*)g;
#endif
}

__device__ inline v8f zero_v8f() {
  v8f z;
#pragma unroll
  for (int i = 0; i < 8; i++) z[i] = 0.0f;
  return z;
}

// D = A(16x32 bf16) x B(32x16 bf16) + C(16x16 f32)
__device__ inline v8f wmma_bf16(v16bf a, v16bf b, v8f c) {
  return __builtin_amdgcn_wmma_f32_16x16x32_bf16(
      false, a, false, b, (short)0, c, false, false);
}

// Fragment = two 16-byte halves at +0 and +32 bytes (K = kb..kb+7, kb+16..kb+23)
__device__ inline v16bf load_frag(const __bf16* p) {
  v8bf lo = *(const v8bf*)p;
  v8bf hi = *(const v8bf*)(p + 16);
  return __builtin_shufflevector(lo, hi, 0, 1, 2, 3, 4, 5, 6, 7,
                                 8, 9, 10, 11, 12, 13, 14, 15);
}

// ---------------------------------------------------------------------------
// Prep kernel A: fp32 -> bf16 flat convert.
// ---------------------------------------------------------------------------
__global__ __launch_bounds__(256) void cvt_bf16(const float* __restrict__ src,
                                                __bf16* __restrict__ dst) {
  size_t i = ((size_t)blockIdx.x * 256 + threadIdx.x) * 4;
  v4f a = *(const v4f*)(src + i);
  v4bf b;
  b[0] = (__bf16)a[0]; b[1] = (__bf16)a[1];
  b[2] = (__bf16)a[2]; b[3] = (__bf16)a[3];
  *(v4bf*)(dst + i) = b;
}

// ---------------------------------------------------------------------------
// Prep kernel B: W[k][n] fp32 -> Wt[n][k] bf16 (tiled transpose through LDS).
// ---------------------------------------------------------------------------
__global__ __launch_bounds__(256) void transpose_w(const float* __restrict__ W,
                                                   __bf16* __restrict__ Wt) {
  __shared__ float t[32][33];
  const int tx = threadIdx.x & 31;
  const int ty = threadIdx.x >> 5;      // 0..7
  const int k0 = blockIdx.y * 32;
  const int n0 = blockIdx.x * 32;
#pragma unroll
  for (int i = 0; i < 4; i++)
    t[ty + 8 * i][tx] = W[(size_t)(k0 + ty + 8 * i) * D_ + n0 + tx];
  __syncthreads();
#pragma unroll
  for (int i = 0; i < 4; i++)
    Wt[(size_t)(n0 + ty + 8 * i) * D_ + k0 + tx] = (__bf16)t[tx][ty + 8 * i];
}

// ---------------------------------------------------------------------------
// Async-staged 128x32 bf16 tile copy: 512 x16B chunks, 256 threads -> 2 each.
// LDS row stride 40 bf16 (80B = 5x16B, conflict-free b128 fragment reads).
// ---------------------------------------------------------------------------
__device__ inline void stage_tile(const __bf16* gsrc, __bf16* lds, int tid) {
#pragma unroll
  for (int it = 0; it < 2; it++) {
    int id = tid + it * 256;
    int r = id >> 2, c = id & 3;
    copy16(gsrc + (size_t)r * D_ + c * 8, lds + r * 40 + c * 8);
  }
}

// ---------------------------------------------------------------------------
// Kernel 1: QKV projection.  C[4096x1024] = Xbf[4096x1024] * Wt[n][k]^T
// Block: 256 thr (8 waves), tile 128x128, wave 32x64 (2x4 C-tiles), K-step 32,
// double-buffered async LDS staging for both A and B.
// transpose_v==0 -> dst[bh][s][d] ; ==1 -> dst[bh][d][s] (for V)
// ---------------------------------------------------------------------------
__global__ __launch_bounds__(256) void qkv_gemm(const __bf16* __restrict__ A,
                                                const __bf16* __restrict__ Bt,
                                                __bf16* __restrict__ dst,
                                                int transpose_v) {
  __shared__ __bf16 ldsA[2][128 * 40];
  __shared__ __bf16 ldsB[2][128 * 40];

  const int tid   = threadIdx.x;
  const int lane  = tid & 31;
  const int ww    = tid >> 5;
  const int m0    = blockIdx.y * 128;
  const int n0    = blockIdx.x * 128;
  const int wm    = (ww & 3) * 32;
  const int wn    = (ww >> 2) * 64;
  const int l15   = lane & 15;
  const int lhalf = lane >> 4;
  const int kb    = lhalf * 8;

  const __bf16* Ag = A  + (size_t)m0 * D_;
  const __bf16* Bg = Bt + (size_t)n0 * D_;

  v8f acc[2][4];
#pragma unroll
  for (int i = 0; i < 2; i++)
#pragma unroll
    for (int j = 0; j < 4; j++) acc[i][j] = zero_v8f();

  stage_tile(Ag, ldsA[0], tid);
  stage_tile(Bg, ldsB[0], tid);
  wait_async();
  __syncthreads();

  for (int kk = 0; kk < D_; kk += 32) {
    const int buf = (kk >> 5) & 1;
    if (kk + 32 < D_) {                       // prefetch next K-step
      stage_tile(Ag + kk + 32, ldsA[buf ^ 1], tid);
      stage_tile(Bg + kk + 32, ldsB[buf ^ 1], tid);
    }

    v16bf af[2], bfg[4];
#pragma unroll
    for (int i = 0; i < 2; i++)
      af[i] = load_frag(&ldsA[buf][(wm + i * 16 + l15) * 40 + kb]);
#pragma unroll
    for (int j = 0; j < 4; j++)
      bfg[j] = load_frag(&ldsB[buf][(wn + j * 16 + l15) * 40 + kb]);
#pragma unroll
    for (int i = 0; i < 2; i++)
#pragma unroll
      for (int j = 0; j < 4; j++)
        acc[i][j] = wmma_bf16(af[i], bfg[j], acc[i][j]);

    wait_async();
    __syncthreads();
  }

  // store head-major bf16
#pragma unroll
  for (int i = 0; i < 2; i++) {
#pragma unroll
    for (int j = 0; j < 4; j++) {
      int ng = n0 + wn + j * 16 + l15;
      int h  = ng >> 6;
      int d  = ng & 63;
#pragma unroll
      for (int vi = 0; vi < 8; vi++) {
        int mg = m0 + wm + i * 16 + lhalf * 8 + vi;
        int b  = mg >> 11;
        int s  = mg & (S_ - 1);
        __bf16 val = (__bf16)acc[i][j][vi];
        if (!transpose_v)
          dst[(((size_t)(b * H_ + h)) * S_ + s) * DA_ + d] = val;
        else
          dst[(((size_t)(b * H_ + h)) * DA_ + d) * S_ + s] = val;
      }
    }
  }
}

// ---------------------------------------------------------------------------
// Kernel 2: causal flash attention.
// Grid: (S/64, B*H). Block 128 thr (4 waves); each wave owns 16 query rows.
// K/V tiles for each 32-key step staged ONCE per block into LDS via async
// copies (double buffered) and shared by all 4 waves.
// LDS strides: K 72 bf16 (144B, 16-way conflict-free), V 40 bf16.
// ---------------------------------------------------------------------------
__global__ __launch_bounds__(128) void flash_attn(const __bf16* __restrict__ Q,
                                                  const __bf16* __restrict__ K,
                                                  const __bf16* __restrict__ Vt,
                                                  __bf16* __restrict__ attn) {
  __shared__ __bf16 ldsK[2][32 * 72];
  __shared__ __bf16 ldsV[2][64 * 40];
  __shared__ __bf16 ldsP[4][16 * 40];

  const int tid   = threadIdx.x;
  const int lane  = tid & 31;
  const int w     = tid >> 5;            // wave 0..3
  const int l15   = lane & 15;
  const int lhalf = lane >> 4;
  const int kb    = lhalf * 8;
  const int bh    = blockIdx.y;
  const int q0b   = blockIdx.x * 64;
  const int q0    = q0b + w * 16;

  const __bf16* Qg = Q  + (size_t)bh * S_ * DA_;
  const __bf16* Kg = K  + (size_t)bh * S_ * DA_;
  const __bf16* Vg = Vt + (size_t)bh * DA_ * S_;
  __bf16* Pl = ldsP[w];

  // resident Q A-fragments (d 0..31, 32..63)
  v16bf aq0, aq1;
  {
    int r = q0 + l15;
    aq0 = load_frag(&Qg[(size_t)r * DA_ + kb]);
    aq1 = load_frag(&Qg[(size_t)r * DA_ + 32 + kb]);
  }

  v8f o[4];
#pragma unroll
  for (int c = 0; c < 4; c++) o[c] = zero_v8f();
  float m_i[8], l_i[8];
#pragma unroll
  for (int i = 0; i < 8; i++) { m_i[i] = -1e30f; l_i[i] = 0.0f; }

  const float sc = 0.125f * 1.44269504088896f;  // (1/sqrt(64)) * log2(e)

  // stage one 32-key K/V tile: K 32x64 (256 chunks) + V 64x32 (256 chunks)
  auto stage_kv = [&](int k0, int buf) {
#pragma unroll
    for (int it = 0; it < 2; it++) {
      int id = tid + it * 128;
      int r = id >> 3, c = id & 7;       // K: 8 chunks per 64-elem row
      copy16(Kg + (size_t)(k0 + r) * DA_ + c * 8, &ldsK[buf][r * 72 + c * 8]);
    }
#pragma unroll
    for (int it = 0; it < 2; it++) {
      int id = tid + it * 128;
      int r = id >> 2, c = id & 3;       // V: 4 chunks per 32-elem row
      copy16(Vg + (size_t)r * S_ + k0 + c * 8, &ldsV[buf][r * 40 + c * 8]);
    }
  };

  const int nkt = blockIdx.x * 2 + 2;    // q0b/32 + 2 causal key tiles
  stage_kv(0, 0);
  wait_async();
  __syncthreads();

  for (int kt = 0; kt < nkt; kt++) {
    const int k0  = kt * 32;
    const int buf = kt & 1;
    if (kt + 1 < nkt) stage_kv(k0 + 32, buf ^ 1);

    if (k0 <= q0 + 15) {                 // this wave still inside causal band
      int key0 = k0 + l15;
      int key1 = k0 + 16 + l15;
      v16bf bk00 = load_frag(&ldsK[buf][(l15) * 72 + kb]);
      v16bf bk01 = load_frag(&ldsK[buf][(l15) * 72 + 32 + kb]);
      v16bf bk10 = load_frag(&ldsK[buf][(16 + l15) * 72 + kb]);
      v16bf bk11 = load_frag(&ldsK[buf][(16 + l15) * 72 + 32 + kb]);

      v8f s0 = zero_v8f(), s1 = zero_v8f();
      s0 = wmma_bf16(aq0, bk00, s0);
      s0 = wmma_bf16(aq1, bk01, s0);
      s1 = wmma_bf16(aq0, bk10, s1);
      s1 = wmma_bf16(aq1, bk11, s1);

#pragma unroll
      for (int i = 0; i < 8; i++) {
        int row = q0 + lhalf * 8 + i;
        float t0 = (key0 <= row) ? s0[i] * sc : -1e30f;
        float t1 = (key1 <= row) ? s1[i] * sc : -1e30f;

        float rmax = fmaxf(t0, t1);
#pragma unroll
        for (int off = 8; off >= 1; off >>= 1)
          rmax = fmaxf(rmax, __shfl_xor(rmax, off, 16));

        float mnew  = fmaxf(m_i[i], rmax);
        float scale = __builtin_amdgcn_exp2f(m_i[i] - mnew);
        m_i[i] = mnew;

        float p0 = __builtin_amdgcn_exp2f(t0 - mnew);
        float p1 = __builtin_amdgcn_exp2f(t1 - mnew);
        float rsum = p0 + p1;
#pragma unroll
        for (int off = 8; off >= 1; off >>= 1)
          rsum += __shfl_xor(rsum, off, 16);
        l_i[i] = l_i[i] * scale + rsum;

#pragma unroll
        for (int c = 0; c < 4; c++) o[c][i] *= scale;

        int prow = lhalf * 8 + i;        // C-layout row within tile
        Pl[prow * 40 + l15]      = (__bf16)p0;
        Pl[prow * 40 + 16 + l15] = (__bf16)p1;
      }

      // P back in A-layout (wave-private LDS; same-wave DS ordering)
      v16bf ap = load_frag(&Pl[l15 * 40 + kb]);

#pragma unroll
      for (int c = 0; c < 4; c++) {
        int d = c * 16 + l15;
        v16bf bv = load_frag(&ldsV[buf][d * 40 + kb]);
        o[c] = wmma_bf16(ap, bv, o[c]);
      }
    }

    wait_async();
    __syncthreads();
  }

  // finalize (divide by row sum) and store bf16, re-interleaving heads
#pragma unroll
  for (int i = 0; i < 8; i++) {
    float inv = 1.0f / l_i[i];
    int row = q0 + lhalf * 8 + i;
    int b = bh >> 4;
    int h = bh & 15;
    size_t base = ((size_t)(b * S_ + row)) * D_ + h * DA_;
#pragma unroll
    for (int c = 0; c < 4; c++)
      attn[base + c * 16 + l15] = (__bf16)(o[c][i] * inv);
  }
}

// ---------------------------------------------------------------------------
// Kernel 3: output projection: out_f32[4096x1024] = At_bf16 * Wto[n][k]^T
// Same tiling/pipelining as kernel 1; fp32 stores to d_out.
// ---------------------------------------------------------------------------
__global__ __launch_bounds__(256) void out_gemm(const __bf16* __restrict__ A,
                                                const __bf16* __restrict__ Bt,
                                                float* __restrict__ out) {
  __shared__ __bf16 ldsA[2][128 * 40];
  __shared__ __bf16 ldsB[2][128 * 40];

  const int tid   = threadIdx.x;
  const int lane  = tid & 31;
  const int ww    = tid >> 5;
  const int m0    = blockIdx.y * 128;
  const int n0    = blockIdx.x * 128;
  const int wm    = (ww & 3) * 32;
  const int wn    = (ww >> 2) * 64;
  const int l15   = lane & 15;
  const int lhalf = lane >> 4;
  const int kb    = lhalf * 8;

  const __bf16* Ag = A  + (size_t)m0 * D_;
  const __bf16* Bg = Bt + (size_t)n0 * D_;

  v8f acc[2][4];
#pragma unroll
  for (int i = 0; i < 2; i++)
#pragma unroll
    for (int j = 0; j < 4; j++) acc[i][j] = zero_v8f();

  stage_tile(Ag, ldsA[0], tid);
  stage_tile(Bg, ldsB[0], tid);
  wait_async();
  __syncthreads();

  for (int kk = 0; kk < D_; kk += 32) {
    const int buf = (kk >> 5) & 1;
    if (kk + 32 < D_) {
      stage_tile(Ag + kk + 32, ldsA[buf ^ 1], tid);
      stage_tile(Bg + kk + 32, ldsB[buf ^ 1], tid);
    }

    v16bf af[2], bfg[4];
#pragma unroll
    for (int i = 0; i < 2; i++)
      af[i] = load_frag(&ldsA[buf][(wm + i * 16 + l15) * 40 + kb]);
#pragma unroll
    for (int j = 0; j < 4; j++)
      bfg[j] = load_frag(&ldsB[buf][(wn + j * 16 + l15) * 40 + kb]);
#pragma unroll
    for (int i = 0; i < 2; i++)
#pragma unroll
      for (int j = 0; j < 4; j++)
        acc[i][j] = wmma_bf16(af[i], bfg[j], acc[i][j]);

    wait_async();
    __syncthreads();
  }

#pragma unroll
  for (int i = 0; i < 2; i++) {
#pragma unroll
    for (int j = 0; j < 4; j++) {
      int ng = n0 + wn + j * 16 + l15;
#pragma unroll
      for (int vi = 0; vi < 8; vi++) {
        int mg = m0 + wm + i * 16 + lhalf * 8 + vi;
        out[(size_t)mg * D_ + ng] = acc[i][j][vi];
      }
    }
  }
}

// ---------------------------------------------------------------------------
// Launch pipeline (all on `stream`):
//   prep: x->bf16, 4x weight transpose->bf16
//   qkv GEMMs (Q, K head-major; V transposed) -> flash attention -> out proj
// Workspace (bf16 elems): Xbf 4Mi | Wtq/Wtk/Wtv/Wto 1Mi each | Qb/Kb/Vt/At 4Mi
// = 48 MiB total.
// ---------------------------------------------------------------------------
extern "C" void kernel_launch(void* const* d_in, const int* in_sizes, int n_in,
                              void* d_out, int out_size, void* d_ws, size_t ws_size,
                              hipStream_t stream) {
  (void)in_sizes; (void)n_in; (void)out_size; (void)ws_size;
  const float* x  = (const float*)d_in[0];
  // d_in[1] = causal mask: recomputed analytically in-kernel
  const float* wq = (const float*)d_in[2];
  const float* wk = (const float*)d_in[3];
  const float* wv = (const float*)d_in[4];
  const float* wo = (const float*)d_in[5];
  float* out = (float*)d_out;

  const size_t XN = (size_t)M_ * D_;             // 4,194,304
  const size_t WN = (size_t)D_ * D_;             // 1,048,576
  __bf16* Xbf = (__bf16*)d_ws;
  __bf16* Wtq = Xbf + XN;
  __bf16* Wtk = Wtq + WN;
  __bf16* Wtv = Wtk + WN;
  __bf16* Wto = Wtv + WN;
  __bf16* Qb  = Wto + WN;
  __bf16* Kb  = Qb + XN;
  __bf16* Vt  = Kb + XN;
  __bf16* At  = Vt + XN;

  // prep
  cvt_bf16<<<dim3((unsigned)(XN / 1024)), dim3(256), 0, stream>>>(x, Xbf);
  dim3 tg(32, 32), tb(256);
  transpose_w<<<tg, tb, 0, stream>>>(wq, Wtq);
  transpose_w<<<tg, tb, 0, stream>>>(wk, Wtk);
  transpose_w<<<tg, tb, 0, stream>>>(wv, Wtv);
  transpose_w<<<tg, tb, 0, stream>>>(wo, Wto);

  // projections
  dim3 gg(D_ / 128, M_ / 128), gb(256);          // (8, 32)
  qkv_gemm<<<gg, gb, 0, stream>>>(Xbf, Wtq, Qb, 0);
  qkv_gemm<<<gg, gb, 0, stream>>>(Xbf, Wtk, Kb, 0);
  qkv_gemm<<<gg, gb, 0, stream>>>(Xbf, Wtv, Vt, 1);

  // attention
  flash_attn<<<dim3(S_ / 64, BH_), dim3(128), 0, stream>>>(Qb, Kb, Vt, At);

  // output projection
  out_gemm<<<gg, gb, 0, stream>>>(At, Wto, out);
}
